// LSTM_layer_51926154609330
// MI455X (gfx1250) — matrix operand order repaired and checked by reading, not compiled, not run on hardware
//
#include <hip/hip_runtime.h>
#include <hip/hip_bf16.h>

// ---------- types ----------
typedef __attribute__((ext_vector_type(16))) __bf16 v16bf;
typedef __attribute__((ext_vector_type(8)))  float  v8f;
typedef unsigned int u32x4 __attribute__((ext_vector_type(4)));
typedef unsigned int u32x8 __attribute__((ext_vector_type(8)));

union Frag { v16bf v; uint4 q[2]; };

__device__ __forceinline__ unsigned short f2bfbits(float f) {
    unsigned u = __builtin_bit_cast(unsigned, f);
    unsigned r = u + 0x7fffu + ((u >> 16) & 1u);   // RNE
    return (unsigned short)(r >> 16);
}

__device__ __forceinline__ float sigm(float x) {
    return 1.0f / (1.0f + __expf(-x));
}
__device__ __forceinline__ float tanh_fast(float x) {
    x = fminf(fmaxf(x, -15.0f), 15.0f);
    float e = __expf(2.0f * x);
    return (e - 1.0f) / (e + 1.0f);
}

// dimensions
#define BATCH 32
#define SEQ   1024
#define DIN   512
#define DH    512
#define NCOL  2048            // 4 gates * 512
#define ROWS  (BATCH * SEQ)   // 32768

// ---------- phase 0a: convert x (fp32) -> bf16 row-major ----------
__global__ __launch_bounds__(256) void cvt_x_kernel(const float* __restrict__ x,
                                                    unsigned short* __restrict__ xbf) {
    size_t idx = (size_t)blockIdx.x * 256 + threadIdx.x;     // one float4 per thread
    float4 v = ((const float4*)x)[idx];
    ushort4 o;
    o.x = f2bfbits(v.x); o.y = f2bfbits(v.y);
    o.z = f2bfbits(v.z); o.w = f2bfbits(v.w);
    ((ushort4*)xbf)[idx] = o;
}

// ---------- phase 0b: pack 4 gate weight matrices [512,512] into WMMA B-fragment order ----------
// pack element o:  kt = o>>16, n = (o>>5)&2047, hi = (o>>4)&1, h = o&15
// source k = kt*32 + (h&8)*2 + hi*8 + (h&7) ; column n -> gate g = n>>9, col c = n&511
__global__ __launch_bounds__(256) void pack_w_kernel(const float* __restrict__ Wg0,
                                                     const float* __restrict__ Wg1,
                                                     const float* __restrict__ Wg2,
                                                     const float* __restrict__ Wg3,
                                                     unsigned short* __restrict__ pack) {
    unsigned o  = blockIdx.x * 256 + threadIdx.x;  // 0 .. 1048575
    unsigned kt = o >> 16;
    unsigned rem = o & 65535u;
    unsigned n  = rem >> 5;
    unsigned j  = rem & 31u;
    unsigned hi = j >> 4;
    unsigned h  = j & 15u;
    unsigned k  = kt * 32 + (h & 8u) * 2 + hi * 8 + (h & 7u);
    unsigned g  = n >> 9;
    unsigned c  = n & 511u;
    const float* W = (g == 0) ? Wg0 : (g == 1) ? Wg1 : (g == 2) ? Wg2 : Wg3;
    pack[o] = f2bfbits(W[(size_t)k * 512 + c]);
}

// ---------- phase 1: gates = x @ Wx + b   (bf16 WMMA, f32 accumulate) ----------
__global__ __launch_bounds__(256) void gates_gemm_kernel(const unsigned short* __restrict__ xbf,
                                                         const unsigned short* __restrict__ wxp,
                                                         const float* __restrict__ bi,
                                                         const float* __restrict__ bff,
                                                         const float* __restrict__ bo,
                                                         const float* __restrict__ bc,
                                                         float* __restrict__ gates) {
    unsigned gid  = blockIdx.x * 8 + (threadIdx.x >> 5);   // global wave id
    unsigned mt   = gid >> 7;          // 0..2047  (M tiles of 16 rows)
    unsigned nt   = gid & 127u;        // 0..127   (N tiles of 16 cols)
    unsigned lane = threadIdx.x & 31u;
    unsigned lo   = lane & 15u;
    unsigned hi   = lane >> 4;

    v8f acc = {0.f, 0.f, 0.f, 0.f, 0.f, 0.f, 0.f, 0.f};

    const unsigned short* arow = xbf + ((size_t)(mt * 16 + lo)) * DIN + hi * 8;
    const unsigned short* bcol = wxp + (size_t)(nt * 16 + lo) * 32 + hi * 16;

#pragma unroll 4
    for (unsigned kt = 0; kt < 16; ++kt) {
        Frag a, b;
        const uint4* pa = (const uint4*)(arow + kt * 32);
        a.q[0] = pa[0];            // k = kbase .. kbase+7
        a.q[1] = pa[2];            // k = kbase+16 .. kbase+23
        const uint4* pb = (const uint4*)(bcol + (size_t)kt * 65536);
        b.q[0] = pb[0];
        b.q[1] = pb[1];
        acc = __builtin_amdgcn_wmma_f32_16x16x32_bf16(false, a.v, false, b.v,
                                                      (short)0, acc, false, false);
    }

    unsigned ng = nt * 16 + lo;          // global column
    unsigned g  = ng >> 9;
    unsigned h  = ng & 511u;
    const float* bias = (g == 0) ? bi : (g == 1) ? bff : (g == 2) ? bo : bc;
    float bv = bias[h];

#pragma unroll
    for (unsigned r = 0; r < 8; ++r) {
        unsigned m = mt * 16 + hi * 8 + r;   // global row = b*SEQ + s
        unsigned bb = m >> 10;
        unsigned ss = m & 1023u;
        gates[(((size_t)ss * 4 + g) * BATCH + bb) * DH + h] = acc[r] + bv;
    }
}

// ---------- phase 2: persistent recurrent kernel ----------
// 8 blocks x 256 threads = 64 waves; wave wi in block: mt = wi&1 (batch tile),
// lht = wi>>1 (local h tile 0..3); global ht = blockIdx*4 + lht.
// Block stages its 256KB Wh slice into LDS once via the Tensor Data Mover,
// then 1024 steps read B fragments from LDS (ds_load_b128).
__global__ __launch_bounds__(256) void lstm_rec_kernel(const unsigned short* __restrict__ whp,
                                                       const float* __restrict__ gates,
                                                       float* __restrict__ cst,
                                                       unsigned short* __restrict__ hbf,   // 2 ping-pong bufs of 32*512
                                                       float* __restrict__ out,
                                                       float* __restrict__ hfin,
                                                       unsigned* __restrict__ syncc) {
    // LDS: [kt 0..15][gate 0..3][local col 0..63][hi 0..1][h 0..15] bf16 = 256 KB
    __shared__ unsigned short bsh[16 * 4 * 64 * 32];

    unsigned wi   = threadIdx.x >> 5;
    unsigned mt   = wi & 1u;
    unsigned lht  = wi >> 1;           // 0..3
    unsigned lane = threadIdx.x & 31u;
    unsigned lo   = lane & 15u;
    unsigned hi   = lane >> 4;
    const unsigned nblk = gridDim.x;

    // ---- one-shot TDM stage of this block's Wh slice into LDS ----
    if (wi == 0) {
        unsigned lds_base = (unsigned)(uintptr_t)(&bsh[0]);
        unsigned long long ga = (unsigned long long)(uintptr_t)whp
                              + (unsigned long long)blockIdx.x * 4096ull;  // 64 cols * 32 el * 2B
        // D# group 0: count=1 | lds_addr | global_addr | type=2
        u32x4 g0;
        g0[0] = 1u;
        g0[1] = lds_base;
        g0[2] = (unsigned)(ga & 0xFFFFFFFFull);
        g0[3] = (unsigned)((ga >> 32) & 0x1FFFFFFull) | (2u << 30);
        // D# group 1: data_size=2B; tensor_dim0=2048, tensor_dim1=4;
        // tile = 2048 x 4 x 16; dim0_stride=16384 el, dim1_stride=65536 el
        u32x8 g1;
        g1[0] = 1u << 16;              // data_size = 1 (2 bytes)
        g1[1] = 2048u << 16;           // tensor_dim0[15:0]
        g1[2] = 4u << 16;              // tensor_dim1[15:0]
        g1[3] = 2048u << 16;           // tile_dim0
        g1[4] = 4u | (16u << 16);      // tile_dim1 | tile_dim2
        g1[5] = 16384u;                // tensor_dim0_stride[31:0]
        g1[6] = 0u;                    // dim0_stride[47:32]=0, dim1_stride[15:0]=0
        g1[7] = 1u;                    // tensor_dim1_stride[47:16] = 1 -> 65536
        // D# group 2: tensor_dim2 = 16; rest 0 (tile_dim3 unused)
        u32x4 g2; g2[0] = 16u; g2[1] = 0u; g2[2] = 0u; g2[3] = 0u;
        u32x4 g3; g3[0] = 0u;  g3[1] = 0u; g3[2] = 0u; g3[3] = 0u;
        asm volatile("tensor_load_to_lds %0, %1, %2, %3"
                     :: "s"(g0), "s"(g1), "s"(g2), "s"(g3)
                     : "memory");
        __builtin_amdgcn_s_wait_tensorcnt(0);
    }
    __syncthreads();

    const unsigned short* bfrag_base = bsh + ((size_t)lht * 16 + lo) * 32 + hi * 16;

    for (unsigned s = 0; s < SEQ; ++s) {
        const unsigned short* rb = hbf + (size_t)(s & 1u) * (BATCH * DH);
        unsigned short*       wb = hbf + (size_t)((s & 1u) ^ 1u) * (BATCH * DH);

        v8f acc0 = {0.f,0.f,0.f,0.f,0.f,0.f,0.f,0.f};
        v8f acc1 = acc0, acc2 = acc0, acc3 = acc0;

        const unsigned short* arow = rb + ((size_t)(mt * 16 + lo)) * DH + hi * 8;

#pragma unroll 2
        for (unsigned kt = 0; kt < 16; ++kt) {
            Frag a;
            const uint4* pa = (const uint4*)(arow + kt * 32);
            a.q[0] = pa[0];
            a.q[1] = pa[2];
            Frag b0, b1, b2, b3;
#pragma unroll
            for (unsigned g = 0; g < 4; ++g) {
                // LDS element offset: ((kt*4+g)*64 + lht*16+lo)*32 + hi*16
                const uint4* pb = (const uint4*)(bfrag_base + ((size_t)kt * 4 + g) * 2048);
                Frag b; b.q[0] = pb[0]; b.q[1] = pb[1];
                if (g == 0) b0 = b; else if (g == 1) b1 = b; else if (g == 2) b2 = b; else b3 = b;
            }
            acc0 = __builtin_amdgcn_wmma_f32_16x16x32_bf16(false, a.v, false, b0.v, (short)0, acc0, false, false);
            acc1 = __builtin_amdgcn_wmma_f32_16x16x32_bf16(false, a.v, false, b1.v, (short)0, acc1, false, false);
            acc2 = __builtin_amdgcn_wmma_f32_16x16x32_bf16(false, a.v, false, b2.v, (short)0, acc2, false, false);
            acc3 = __builtin_amdgcn_wmma_f32_16x16x32_bf16(false, a.v, false, b3.v, (short)0, acc3, false, false);
        }

        // fused LSTM pointwise for this wave's 16x16 (b,h) tile
        unsigned h = (blockIdx.x * 4 + lht) * 16 + lo;
        size_t gbase = ((size_t)s * 4) * (BATCH * DH);
#pragma unroll
        for (unsigned r = 0; r < 8; ++r) {
            unsigned m  = hi * 8 + r;
            unsigned bb = mt * 16 + m;
            size_t gi_idx = gbase + (size_t)bb * DH + h;
            float gi = acc0[r] + gates[gi_idx];
            float gf = acc1[r] + gates[gi_idx + 1 * BATCH * DH];
            float go = acc2[r] + gates[gi_idx + 2 * BATCH * DH];
            float gc = acc3[r] + gates[gi_idx + 3 * BATCH * DH];
            float I = sigm(gi), F = sigm(gf), O = sigm(go);
            float Cd = tanh_fast(gc);
            size_t cidx = (size_t)bb * DH + h;
            float Cn = F * cst[cidx] + I * Cd;
            float Hn = O * tanh_fast(Cn);
            cst[cidx] = Cn;
            wb[cidx]  = f2bfbits(Hn);
            out[((size_t)bb * SEQ + s) * DH + h] = Hn;
            if (s == SEQ - 1) hfin[cidx] = Hn;
        }

        // device-wide barrier: all H writes visible before next step reads them
        __syncthreads();
        __threadfence();
        if (threadIdx.x == 0) {
            __hip_atomic_fetch_add(syncc, 1u, __ATOMIC_ACQ_REL, __HIP_MEMORY_SCOPE_AGENT);
            unsigned target = (s + 1) * nblk;
            while (__hip_atomic_load(syncc, __ATOMIC_ACQUIRE, __HIP_MEMORY_SCOPE_AGENT) < target) {
                __builtin_amdgcn_s_sleep(1);
            }
        }
        __syncthreads();
    }
}

// ---------- launch ----------
extern "C" void kernel_launch(void* const* d_in, const int* in_sizes, int n_in,
                              void* d_out, int out_size, void* d_ws, size_t ws_size,
                              hipStream_t stream) {
    const float* x   = (const float*)d_in[0];
    const float* Wxi = (const float*)d_in[1];
    const float* Whi = (const float*)d_in[2];
    const float* bi  = (const float*)d_in[3];
    const float* Wxf = (const float*)d_in[4];
    const float* Whf = (const float*)d_in[5];
    const float* bff = (const float*)d_in[6];
    const float* Wxo = (const float*)d_in[7];
    const float* Who = (const float*)d_in[8];
    const float* bo  = (const float*)d_in[9];
    const float* Wxc = (const float*)d_in[10];
    const float* Whc = (const float*)d_in[11];
    const float* bc  = (const float*)d_in[12];

    char* ws = (char*)d_ws;
    float*          gates = (float*)ws;                                  // 268,435,456 B
    unsigned short* xbf   = (unsigned short*)(ws + 268435456);           //  33,554,432 B
    unsigned short* wxp   = (unsigned short*)(ws + 301989888);           //   2,097,152 B
    unsigned short* whp   = (unsigned short*)(ws + 304087040);           //   2,097,152 B
    float*          cst   = (float*)(ws + 306184192);                    //      65,536 B
    unsigned short* hbf   = (unsigned short*)(ws + 306249728);           //      65,536 B (2 bufs)
    unsigned*       syncc = (unsigned*)(ws + 306315264);                 //         256 B

    hipMemsetAsync(cst,   0, 65536, stream);   // C0 = 0
    hipMemsetAsync(hbf,   0, 65536, stream);   // H0 = 0 (both ping-pong bufs)
    hipMemsetAsync(syncc, 0, 256,   stream);   // barrier counter

    cvt_x_kernel<<<16384, 256, 0, stream>>>(x, xbf);
    pack_w_kernel<<<4096, 256, 0, stream>>>(Wxi, Wxf, Wxo, Wxc, wxp);
    pack_w_kernel<<<4096, 256, 0, stream>>>(Whi, Whf, Who, Whc, whp);
    gates_gemm_kernel<<<32768, 256, 0, stream>>>(xbf, wxp, bi, bff, bo, bc, gates);

    float* out  = (float*)d_out;
    float* hfin = out + (size_t)BATCH * SEQ * DH;
    lstm_rec_kernel<<<8, 256, 0, stream>>>(whp, gates, cst, hbf, out, hfin, syncc);
}
